// MambaAdapter_14920716386472
// MI455X (gfx1250) — compile-verified, matrix operand order checked by reading, not run on hardware
//
#include <hip/hip_runtime.h>
#include <hip/hip_bf16.h>
#include <math.h>

// ---------------------------------------------------------------------------
// Mamba block forward: B=1, d_model=256, d_inner=512, d_state=16, dt_rank=16,
// d_conv=4, L = 8*32*32 = 8192.
//
// WMMA operands live in "fragment-major" swizzled layout: each 16(M/N)x32(K)
// f16 tile stored as 32 lanes x 16 contiguous halfs (ISA 7.12.2 layout
// inverted) -> fragment load = one v16h load = 2x global_load_b128.
// B/A fragments are batch-loaded (8 at a time) ahead of each 8-WMMA chain so
// load latency overlaps the matrix pipe.
// ---------------------------------------------------------------------------

#define LSEQ 8192
#define DMODEL 256
#define DINNER 512
#define DSTATE 16
#define DTRANK 16
#define NCHUNK 32
#define CHUNK 256

typedef __attribute__((ext_vector_type(16))) _Float16 v16h;
typedef __attribute__((ext_vector_type(8)))  float    v8f;

// K offset inside a 16-bit WMMA operand for VGPR-pair v (0..7), half = lane>>4
__device__ __forceinline__ int wmma_k(int v, int hf) {
    return ((v >> 2) << 4) + hf * 8 + ((v & 3) << 1);
}

// Inverse mapping: element (r, kk) of a 16x32 tile -> offset (0..511) inside
// the fragment-major tile (lane*16 + slot).  r = lane dim (M for A, N for B).
__device__ __forceinline__ int frag_off(int r, int kk) {
    int g = kk >> 4, rem = kk & 15;
    int hf = rem >> 3, vv = (rem & 7) >> 1, b = kk & 1;
    int slot = (((g << 2) + vv) << 1) + b;
    int lane = (r & 15) | (hf << 4);
    return lane * 16 + slot;
}

// Fragment load from swizzled storage: one 32-byte vector load per lane.
__device__ __forceinline__ v16h load_frag(const _Float16* base, int fragIdx) {
    int lane = threadIdx.x & 31;
    return *(const v16h*)(base + (size_t)fragIdx * 512 + lane * 16);
}

// A fragment from a row-major f32 source: elem(m,k) = p[(m0+m)*ld + k], K masked
// (only used for the tiny dt_proj A tile, amortized over 32 WMMAs)
__device__ __forceinline__ v16h load_tileA_rm_f32(const float* p, int ld, int m0,
                                                  int k0, int kmax) {
    int lane = threadIdx.x & 31, r = lane & 15, hf = lane >> 4;
    v16h t;
#pragma unroll
    for (int v = 0; v < 8; ++v) {
        int k = k0 + wmma_k(v, hf);
        t[2 * v]     = (k < kmax)     ? (_Float16)p[(size_t)(m0 + r) * ld + k]     : (_Float16)0.f;
        t[2 * v + 1] = (k + 1 < kmax) ? (_Float16)p[(size_t)(m0 + r) * ld + k + 1] : (_Float16)0.f;
    }
    return t;
}

__device__ __forceinline__ v8f wmma16(v16h a, v16h b, v8f c) {
    return __builtin_amdgcn_wmma_f32_16x16x32_f16(false, a, false, b, (short)0, c, false, false);
}

// ---------------------------------------------------------------------------
// prep: swizzle+convert weights to fragment-major f16, compute A = -exp(A_log)
__global__ void k_prep(const float* w1, const float* wx, const float* wdt,
                       const float* wo, const float* alog,
                       _Float16* W1S, _Float16* WXS, _Float16* WDS,
                       _Float16* WOS, float* ANEG) {
    int i = blockIdx.x * blockDim.x + threadIdx.x;
    // in_proj B: tiles (nt=e/16: 64) x (kt=c/32: 8)
    if (i < 1024 * 256) {
        int e = i >> 8, c = i & 255;
        int frag = (e >> 4) * 8 + (c >> 5);
        W1S[frag * 512 + frag_off(e & 15, c & 31)] = (_Float16)w1[e * 256 + c];
    }
    // out_proj A: tiles (mt=e/16: 16) x (kt=d/32: 16)
    if (i < 256 * 512) {
        int e = i >> 9, d = i & 511;
        int frag = (e >> 4) * 16 + (d >> 5);
        WOS[frag * 512 + frag_off(e & 15, d & 31)] = (_Float16)wo[e * 512 + d];
    }
    // x_proj B: tiles (nt=j/16: 3) x (kt=d/32: 16)
    if (i < 48 * 512) {
        int j = i >> 9, d = i & 511;
        int frag = (j >> 4) * 16 + (d >> 5);
        WXS[frag * 512 + frag_off(j & 15, d & 31)] = (_Float16)wx[j * 512 + d];
    }
    // dt_proj B: tiles (nt=d/16: 32) x (kt: 1, K padded 16->32 with zeros)
    if (i < 32 * 512) {
        int kk = i >> 9, d = i & 511;
        float v = (kk < DTRANK) ? wdt[d * 16 + kk] : 0.f;
        WDS[(d >> 4) * 512 + frag_off(d & 15, kk)] = (_Float16)v;
    }
    if (i < 512 * 16) ANEG[i] = -__expf(alog[i]);
}

// swizzle raw input x [C][L] f32 -> in_proj A fragments (tiles mt=l/16, kt=c/32)
__global__ void k_swz_x(const float* x, _Float16* XS) {
    int i = blockIdx.x * blockDim.x + threadIdx.x;   // c*8192 + l
    int c = i >> 13, l = i & 8191;
    int frag = (l >> 4) * 8 + (c >> 5);
    XS[frag * 512 + frag_off(l & 15, c & 31)] = (_Float16)x[i];
}

// in_proj: XZ[l][e] = sum_c x[l][c] * w1[e][c]   (e in 0..1023; xin=0..511, z=512..1023)
__global__ void k_inproj(const _Float16* XS, const _Float16* W1S, float* XZ) {
    int wave = (blockIdx.x * blockDim.x + threadIdx.x) >> 5;   // 0..511 M-tiles
    int m0 = wave * 16;
    v16h A[8];
#pragma unroll
    for (int kt = 0; kt < 8; ++kt) A[kt] = load_frag(XS, wave * 8 + kt);
    int lane = threadIdx.x & 31, n = lane & 15, hf = lane >> 4;
    for (int nt = 0; nt < 64; ++nt) {
        if (nt < 63) __builtin_prefetch(W1S + (size_t)(nt + 1) * 8 * 512, 0, 1);
        v16h B[8];
#pragma unroll
        for (int kt = 0; kt < 8; ++kt) B[kt] = load_frag(W1S, nt * 8 + kt);
        v8f acc = {};
#pragma unroll
        for (int kt = 0; kt < 8; ++kt) acc = wmma16(A[kt], B[kt], acc);
        int e = nt * 16 + n;
#pragma unroll
        for (int v = 0; v < 8; ++v)
            XZ[(size_t)(m0 + v + 8 * hf) * 1024 + e] = acc[v];
    }
}

// depthwise causal conv (d_conv=4) + bias + SiLU on xin (= XZ cols 0..511)
// -> U (f32 [L][512]) and swizzled x_proj A fragments UTS (tiles mt=l/16, kt=d/32)
__global__ void k_conv(const float* XZ, const float* cw, const float* cb,
                       float* U, _Float16* UTS) {
    int i = blockIdx.x * blockDim.x + threadIdx.x;   // l*512 + d
    int l = i >> 9, d = i & 511;
    float acc = cb[d];
#pragma unroll
    for (int j = 0; j < 4; ++j) {
        int ll = l - 3 + j;
        if (ll >= 0) acc += cw[d * 4 + j] * XZ[(size_t)ll * 1024 + d];
    }
    float u = acc / (1.0f + __expf(-acc));
    U[i] = u;
    int frag = (l >> 4) * 16 + (d >> 5);
    UTS[frag * 512 + frag_off(l & 15, d & 31)] = (_Float16)u;
}

// x_proj: DBC[l][j] = sum_d u[l][d] * wx[j][d]   (j in 0..47)
__global__ void k_xproj(const _Float16* UTS, const _Float16* WXS, float* DBC) {
    int wave = (blockIdx.x * blockDim.x + threadIdx.x) >> 5;
    int m0 = wave * 16;
    int lane = threadIdx.x & 31, n = lane & 15, hf = lane >> 4;
    v16h A[16];
#pragma unroll
    for (int kt = 0; kt < 16; ++kt) A[kt] = load_frag(UTS, wave * 16 + kt);
    for (int nt = 0; nt < 3; ++nt) {
        v8f acc = {};
#pragma unroll
        for (int half = 0; half < 2; ++half) {
            v16h B[8];
#pragma unroll
            for (int kt = 0; kt < 8; ++kt) B[kt] = load_frag(WXS, nt * 16 + half * 8 + kt);
#pragma unroll
            for (int kt = 0; kt < 8; ++kt) acc = wmma16(A[half * 8 + kt], B[kt], acc);
        }
        int j = nt * 16 + n;
#pragma unroll
        for (int v = 0; v < 8; ++v)
            DBC[(size_t)(m0 + v + 8 * hf) * 48 + j] = acc[v];
    }
}

// dt_proj + bias + softplus:  DT[l][d] = softplus(sum_r DBC[l][r] * wdt[d][r] + b[d])
__global__ void k_dtproj(const float* DBC, const _Float16* WDS, const float* dtb, float* DT) {
    int wave = (blockIdx.x * blockDim.x + threadIdx.x) >> 5;
    int m0 = wave * 16;
    int lane = threadIdx.x & 31, n = lane & 15, hf = lane >> 4;
    v16h A = load_tileA_rm_f32(DBC, 48, m0, 0, DTRANK);   // K padded 16->32
    for (int nt = 0; nt < 32; ++nt) {
        v8f acc = {};
        acc = wmma16(A, load_frag(WDS, nt), acc);
        int d = nt * 16 + n;
        float bias = dtb[d];
#pragma unroll
        for (int v = 0; v < 8; ++v) {
            float xv = acc[v] + bias;
            float sp = (xv > 20.f) ? xv : log1pf(__expf(xv));
            DT[(size_t)(m0 + v + 8 * hf) * DINNER + d] = sp;
        }
    }
}

// scan pass A: per (chunk, d, s) local scan from h=0; emit chunk product P and final F
__global__ void k_scanA(const float* DT, const float* U, const float* DBC,
                        const float* ANEG, float* P, float* F) {
    __shared__ float sdt[CHUNK * 16], su[CHUNK * 16], sB[CHUNK * 16];
    int cg = blockIdx.x;          // channel group 0..31 (16 channels each)
    int ch = blockIdx.y;          // chunk 0..31
    int d0 = cg * 16, l0 = ch * CHUNK;
    int tid = threadIdx.x;
    for (int it = 0; it < 16; ++it) {
        int idx = it * 256 + tid, l = idx >> 4, q = idx & 15;
        sdt[idx] = DT[(size_t)(l0 + l) * DINNER + d0 + q];
        su[idx]  = U[(size_t)(l0 + l) * DINNER + d0 + q];
        sB[idx]  = DBC[(size_t)(l0 + l) * 48 + DTRANK + q];
    }
    __syncthreads();
    int dl = tid >> 4, s = tid & 15;
    int d = d0 + dl;
    float a = ANEG[d * DSTATE + s];
    float h = 0.f, p = 1.f;
    for (int l = 0; l < CHUNK; ++l) {
        float dtv = sdt[l * 16 + dl];
        float da = __expf(dtv * a);
        h = da * h + dtv * sB[l * 16 + s] * su[l * 16 + dl];
        p *= da;
    }
    size_t o = ((size_t)ch * DINNER + d) * DSTATE + s;
    P[o] = p; F[o] = h;
}

// scan pass B: sequential combine of 32 chunk summaries -> per-chunk init states HI
__global__ void k_scanB(const float* P, const float* F, float* HI) {
    int i = blockIdx.x * blockDim.x + threadIdx.x;   // (d*16+s), 8192 total
    float h = 0.f;
    for (int c = 0; c < NCHUNK; ++c) {
        size_t o = (size_t)c * (DINNER * DSTATE) + i;
        HI[o] = h;
        h = P[o] * h + F[o];
    }
}

// scan pass C: replay with correct h_init, reduce over 16 states (shfl_xor, wave32),
// add skip D term, gate with SiLU(z), scatter y into out_proj B fragments
__global__ void k_scanC(const float* DT, const float* U, const float* DBC,
                        const float* ANEG, const float* HI, const float* XZ,
                        const float* Dvec, _Float16* YGS) {
    __shared__ float sdt[CHUNK * 16], sB[CHUNK * 16], sC[CHUNK * 16];
    int cg = blockIdx.x, ch = blockIdx.y;
    int d0 = cg * 16, l0 = ch * CHUNK;
    int tid = threadIdx.x;
    for (int it = 0; it < 16; ++it) {
        int idx = it * 256 + tid, l = idx >> 4, q = idx & 15;
        sdt[idx] = DT[(size_t)(l0 + l) * DINNER + d0 + q];
        sB[idx]  = DBC[(size_t)(l0 + l) * 48 + DTRANK + q];
        sC[idx]  = DBC[(size_t)(l0 + l) * 48 + DTRANK + DSTATE + q];
    }
    __syncthreads();
    int dl = tid >> 4, s = tid & 15;
    int d = d0 + dl;
    float a = ANEG[d * DSTATE + s];
    float h = HI[((size_t)ch * DINNER + d) * DSTATE + s];
    float Dd = Dvec[d];
    int fragk = d >> 5, doffk = d & 31;
    for (int l = 0; l < CHUNK; ++l) {
        float dtv = sdt[l * 16 + dl];
        float uv = U[(size_t)(l0 + l) * DINNER + d];
        float da = __expf(dtv * a);
        h = da * h + dtv * sB[l * 16 + s] * uv;
        float part = h * sC[l * 16 + s];
        part += __shfl_xor(part, 1, 32);
        part += __shfl_xor(part, 2, 32);
        part += __shfl_xor(part, 4, 32);
        part += __shfl_xor(part, 8, 32);
        if (s == 0) {
            int gl = l0 + l;
            float zv = XZ[(size_t)gl * 1024 + DINNER + d];   // z half of xz
            float y = part + uv * Dd;
            y *= zv / (1.f + __expf(-zv));
            int frag = (gl >> 4) * 16 + fragk;               // B tiles: nt=l/16, kt=d/32
            YGS[frag * 512 + frag_off(gl & 15, doffk)] = (_Float16)y;
        }
    }
}

// out_proj (transposed): out[e][l] = sum_d wo[e][d] * y[l][d]
__global__ void k_outproj(const _Float16* WOS, const _Float16* YGS, float* out) {
    int wave = (blockIdx.x * blockDim.x + threadIdx.x) >> 5;  // 0..511 N(l)-tiles
    int lane = threadIdx.x & 31, n = lane & 15, hf = lane >> 4;
    v16h B[16];
#pragma unroll
    for (int kt = 0; kt < 16; ++kt) B[kt] = load_frag(YGS, wave * 16 + kt);
    for (int mt = 0; mt < 16; ++mt) {
        if (mt < 15) __builtin_prefetch(WOS + (size_t)(mt + 1) * 16 * 512, 0, 1);
        v8f acc = {};
#pragma unroll
        for (int half = 0; half < 2; ++half) {
            v16h Af[8];
#pragma unroll
            for (int kt = 0; kt < 8; ++kt) Af[kt] = load_frag(WOS, mt * 16 + half * 8 + kt);
#pragma unroll
            for (int kt = 0; kt < 8; ++kt) acc = wmma16(Af[kt], B[half * 8 + kt], acc);
        }
#pragma unroll
        for (int v = 0; v < 8; ++v) {
            int e = mt * 16 + v + 8 * hf;
            out[(size_t)e * LSEQ + wave * 16 + n] = acc[v];
        }
    }
}

// ---------------------------------------------------------------------------
// workspace layout (bytes); every section is >=1KB aligned so v16h loads are
// 32B-aligned.
static constexpr size_t OFF_W1S  = 0;
static constexpr size_t OFF_WXS  = OFF_W1S  + (size_t)256 * 1024 * 2;    // 512 KB
static constexpr size_t OFF_WDS  = OFF_WXS  + (size_t)512 * 48 * 2;      // 48 KB
static constexpr size_t OFF_WOS  = OFF_WDS  + (size_t)32 * 512 * 2;      // 32 KB
static constexpr size_t OFF_ANEG = OFF_WOS  + (size_t)512 * 256 * 2;     // 256 KB
static constexpr size_t OFF_XS   = OFF_ANEG + (size_t)512 * 16 * 4;      // 32 KB
static constexpr size_t OFF_XZ   = OFF_XS   + (size_t)LSEQ * DMODEL * 2; // 4 MB
static constexpr size_t OFF_U    = OFF_XZ   + (size_t)LSEQ * 1024 * 4;   // 32 MB
static constexpr size_t OFF_UTS  = OFF_U    + (size_t)LSEQ * DINNER * 4; // 16 MB
static constexpr size_t OFF_DBC  = OFF_UTS  + (size_t)LSEQ * DINNER * 2; // 8 MB
static constexpr size_t OFF_DT   = OFF_DBC  + (size_t)LSEQ * 48 * 4;
static constexpr size_t OFF_P    = OFF_DT   + (size_t)LSEQ * DINNER * 4;
static constexpr size_t OFF_F    = OFF_P    + (size_t)NCHUNK * DINNER * DSTATE * 4;
static constexpr size_t OFF_HI   = OFF_F    + (size_t)NCHUNK * DINNER * DSTATE * 4;
static constexpr size_t OFF_YGS  = OFF_HI   + (size_t)NCHUNK * DINNER * DSTATE * 4;
// total ~ 89 MB

extern "C" void kernel_launch(void* const* d_in, const int* in_sizes, int n_in,
                              void* d_out, int out_size, void* d_ws, size_t ws_size,
                              hipStream_t stream) {
    (void)in_sizes; (void)n_in; (void)out_size; (void)ws_size;
    const float* x     = (const float*)d_in[0];   // [256][8192]
    const float* w_in  = (const float*)d_in[1];   // [1024][256]
    const float* convw = (const float*)d_in[2];   // [512][1][4]
    const float* convb = (const float*)d_in[3];   // [512]
    const float* w_x   = (const float*)d_in[4];   // [48][512]
    const float* w_dt  = (const float*)d_in[5];   // [512][16]
    const float* b_dt  = (const float*)d_in[6];   // [512]
    const float* alog  = (const float*)d_in[7];   // [512][16]
    const float* Dvec  = (const float*)d_in[8];   // [512]
    const float* w_out = (const float*)d_in[9];   // [256][512]
    float* out = (float*)d_out;                   // [256][8192]

    char* ws = (char*)d_ws;
    _Float16* W1S = (_Float16*)(ws + OFF_W1S);
    _Float16* WXS = (_Float16*)(ws + OFF_WXS);
    _Float16* WDS = (_Float16*)(ws + OFF_WDS);
    _Float16* WOS = (_Float16*)(ws + OFF_WOS);
    float*    ANEG = (float*)(ws + OFF_ANEG);
    _Float16* XS   = (_Float16*)(ws + OFF_XS);
    float*    XZ   = (float*)(ws + OFF_XZ);
    float*    U    = (float*)(ws + OFF_U);
    _Float16* UTS  = (_Float16*)(ws + OFF_UTS);
    float*    DBC  = (float*)(ws + OFF_DBC);
    float*    DT   = (float*)(ws + OFF_DT);
    float*    P    = (float*)(ws + OFF_P);
    float*    F    = (float*)(ws + OFF_F);
    float*    HI   = (float*)(ws + OFF_HI);
    _Float16* YGS  = (_Float16*)(ws + OFF_YGS);

    k_prep<<<1024, 256, 0, stream>>>(w_in, w_x, w_dt, w_out, alog, W1S, WXS, WDS, WOS, ANEG);
    k_swz_x<<<(DMODEL * LSEQ) / 256, 256, 0, stream>>>(x, XS);
    k_inproj<<<64, 256, 0, stream>>>(XS, W1S, XZ);
    k_conv<<<(LSEQ * DINNER) / 256, 256, 0, stream>>>(XZ, convw, convb, U, UTS);
    k_xproj<<<64, 256, 0, stream>>>(UTS, WXS, DBC);
    k_dtproj<<<64, 256, 0, stream>>>(DBC, WDS, b_dt, DT);
    k_scanA<<<dim3(32, 32), 256, 0, stream>>>(DT, U, DBC, ANEG, P, F);
    k_scanB<<<32, 256, 0, stream>>>(P, F, HI);
    k_scanC<<<dim3(32, 32), 256, 0, stream>>>(DT, U, DBC, ANEG, HI, XZ, Dvec, YGS);
    k_outproj<<<64, 256, 0, stream>>>(WOS, YGS, out);
}
